// GIIF_49761491091720
// MI455X (gfx1250) — matrix operand order repaired and checked by reading, not compiled, not run on hardware
//
#include <hip/hip_runtime.h>
#include <hip/hip_bf16.h>

// ---------------- problem constants (from reference setup_inputs) -----------
#define BSZ   2
#define CCH   128
#define HF    128
#define WF    128
#define HL    64
#define WL    64
#define QN    65536
#define NQ    (BSZ * QN)        // 131072 query rows
#define CHUNK 32768             // rows per pass
#define NCHK  (NQ / CHUNK)      // 4
#define HID   256
#define K0ACT 262
#define K0PAD 288               // 262 padded to multiple of 32

typedef _Float16 half_t;
typedef __attribute__((ext_vector_type(16))) _Float16 v16h;
typedef __attribute__((ext_vector_type(8)))  _Float16 v8h;
typedef __attribute__((ext_vector_type(8)))  float    v8f;

union V16 { v16h v; v8h h[2]; };

// ---------------- workspace layout (bytes) ----------------------------------
#define OFF_WT0  0u
#define OFF_WT1  (OFF_WT0 + 256u * K0PAD * 2u)            // 147456
#define OFF_WT2  (OFF_WT1 + 256u * 256u * 2u)
#define OFF_WT3  (OFF_WT2 + 256u * 256u * 2u)
#define OFF_WGT  (OFF_WT3 + 256u * 256u * 2u)             // 540672
#define OFF_XIN  (OFF_WGT + 4u * NQ * 4u)                 // + 2 MB
#define OFF_ACTA (OFF_XIN + (size_t)CHUNK * K0PAD * 2u)   // + 18.9 MB
#define OFF_ACTB (OFF_ACTA + (size_t)CHUNK * HID * 2u)    // + 16.8 MB
// total ~55 MB

// ---------------- weight convert: W[k][256] f32 -> WT[n][kpad] f16 ----------
__global__ void wcvt_kernel(const float* __restrict__ W, half_t* __restrict__ WT,
                            int kact, int kpad) {
    int n = blockIdx.x;                       // 0..255
    for (int k = threadIdx.x; k < kpad; k += blockDim.x) {
        float v = (k < kact) ? W[(size_t)k * 256 + n] : 0.f;
        WT[(size_t)n * kpad + k] = (half_t)v;
    }
}

// ---------------- per-query blend weights (diagonally swapped areas) --------
__global__ void wgt_kernel(const float* __restrict__ coord, float* __restrict__ wgt) {
    int n = blockIdx.x * blockDim.x + threadIdx.x;
    if (n >= NQ) return;
    float cy = coord[2 * n + 0], cx = coord[2 * n + 1];
    float area[4];
#pragma unroll
    for (int s = 0; s < 4; ++s) {
        float vx = (s >> 1) ? 1.f : -1.f;     // s order: (-1,-1),(-1,1),(1,-1),(1,1)
        float vy = (s & 1) ? 1.f : -1.f;
        float sy = cy + vx * (1.f / HF) + 1e-6f;
        float sx = cx + vy * (1.f / WF) + 1e-6f;
        sy = fminf(fmaxf(sy, -1.f + 1e-6f), 1.f - 1e-6f);
        sx = fminf(fmaxf(sx, -1.f + 1e-6f), 1.f - 1e-6f);
        int iy = (int)floorf(0.5f * (sy + 1.f) * HF); iy = min(max(iy, 0), HF - 1);
        int ix = (int)floorf(0.5f * (sx + 1.f) * WF); ix = min(max(ix, 0), WF - 1);
        float pcy = -1.f + (2.f * iy + 1.f) * (1.f / HF);
        float pcx = -1.f + (2.f * ix + 1.f) * (1.f / WF);
        float r0 = (cy - pcy) * HF, r1 = (cx - pcx) * WF;
        area[s] = fabsf(r0 * r1) + 1e-9f;
    }
    float tot = area[0] + area[1] + area[2] + area[3];
#pragma unroll
    for (int s = 0; s < 4; ++s)
        wgt[(size_t)s * NQ + n] = area[3 - s] / tot;   // swap 0<->3, 1<->2
}

// ---------------- gather + build padded f16 input rows ----------------------
// one wave (32 lanes) per row; 8 rows per 256-thread block
__global__ __launch_bounds__(256)
void build_kernel(const float* __restrict__ feat, const float* __restrict__ lrf,
                  const float* __restrict__ coord, const float* __restrict__ cell,
                  half_t* __restrict__ Xin, int qc, int s) {
    int wave = threadIdx.x >> 5, lane = threadIdx.x & 31;
    int row  = blockIdx.x * 8 + wave;          // local row in chunk
    int n    = qc * CHUNK + row;               // global query row
    int b    = n >> 16;                        // n / QN
    float cy = coord[2 * n + 0], cx = coord[2 * n + 1];
    float vx = (s >> 1) ? 1.f : -1.f;
    float vy = (s & 1) ? 1.f : -1.f;
    float sy = fminf(fmaxf(cy + vx * (1.f / HF) + 1e-6f, -1.f + 1e-6f), 1.f - 1e-6f);
    float sx = fminf(fmaxf(cx + vy * (1.f / WF) + 1e-6f, -1.f + 1e-6f), 1.f - 1e-6f);
    int iy  = min(max((int)floorf(0.5f * (sy + 1.f) * HF), 0), HF - 1);
    int ix  = min(max((int)floorf(0.5f * (sx + 1.f) * WF), 0), WF - 1);
    int liy = min(max((int)floorf(0.5f * (sy + 1.f) * HL), 0), HL - 1);
    int lix = min(max((int)floorf(0.5f * (sx + 1.f) * WL), 0), WL - 1);

    size_t base = (size_t)row * K0PAD;
    size_t fpix = (size_t)(b * CCH) * (HF * WF) + (size_t)iy * WF + ix;
    size_t lpix = (size_t)(b * CCH) * (HL * WL) + (size_t)liy * WL + lix;
#pragma unroll
    for (int i = 0; i < 4; ++i) {
        int cc = lane + i * 32;
        Xin[base + cc]       = (half_t)feat[fpix + (size_t)cc * (HF * WF)];
        Xin[base + 130 + cc] = (half_t)lrf[lpix + (size_t)cc * (HL * WL)];
    }
    if (lane == 0) {
        float pcy = -1.f + (2.f * iy + 1.f) * (1.f / HF);
        float pcx = -1.f + (2.f * ix + 1.f) * (1.f / WF);
        Xin[base + 128] = (half_t)((cy - pcy) * HF);
        Xin[base + 129] = (half_t)((cx - pcx) * WF);
    } else if (lane == 1) {
        Xin[base + 258] = (half_t)(-1.f + (2.f * liy + 1.f) * (1.f / HL));
        Xin[base + 259] = (half_t)(-1.f + (2.f * lix + 1.f) * (1.f / WL));
        Xin[base + 260] = (half_t)(cell[2 * n + 0] * HF);
        Xin[base + 261] = (half_t)(cell[2 * n + 1] * WF);
    } else if (lane >= 2 && lane < 28) {       // zero pad 262..287
        Xin[base + 262 + (lane - 2)] = (half_t)0.f;
    }
}

// ---------------- WMMA GEMM layer: Y = relu(X[rows][kpad] * WT^T + bias) ----
// block = 256 threads = 8 waves; block covers 64 rows x 256 cols.
// wave (strip = w>>2, nq = w&3): rows [blk*64+strip*32, +32), cols [nq*64, +64)
//   -> 2 M-subtiles x 4 N-tiles = 8 WMMA accumulators per wave.
// Weight K-slices (256N x 32K f16 = 16 KB) double-buffered in LDS; next slice's
// global fetch and next A fragments are issued before the current WMMA chain.
// All 4 B fragments are loaded from LDS first (one dscnt wait per 8 WMMAs).
// Epilogue stages the 64x256 f16 tile in LDS, then writes coalesced b128s.
__global__ __launch_bounds__(256)
void gemm_kernel(const half_t* __restrict__ X, const half_t* __restrict__ WT,
                 const float* __restrict__ bias, half_t* __restrict__ Y,
                 int kpad, int ksteps) {
    __shared__ half_t lw[2][256 * 32];          // 2 x 16 KB (reused by epilogue)

    int tid  = threadIdx.x;
    int wave = tid >> 5, lane = tid & 31;
    int strip = wave >> 2;                      // 0..1
    int nq    = wave & 3;                       // 0..3
    int m16   = lane & 15, hf = lane >> 4;
    int rowBase = blockIdx.x * 64 + strip * 32;

    const half_t* arow0 = X + (size_t)(rowBase + m16) * kpad;
    const half_t* arow1 = X + (size_t)(rowBase + 16 + m16) * kpad;
    const half_t* wsrc  = WT + (size_t)tid * kpad;   // thread copies weight row n=tid

    // ---- preload weight slice 0 into LDS buffer 0 and A frags for kk=0 ----
    {
        const v8h* sp = (const v8h*)wsrc;
        v8h* dp = (v8h*)&lw[0][tid * 32];
        dp[0] = sp[0]; dp[1] = sp[1]; dp[2] = sp[2]; dp[3] = sp[3];
    }
    V16 A0, A1;
    A0.h[0] = *(const v8h*)(arow0 + 8 * hf);
    A0.h[1] = *(const v8h*)(arow0 + 16 + 8 * hf);
    A1.h[0] = *(const v8h*)(arow1 + 8 * hf);
    A1.h[1] = *(const v8h*)(arow1 + 16 + 8 * hf);
    __syncthreads();

    v8f acc[2][4] = {};
    for (int kk = 0; kk < ksteps; ++kk) {
        int buf = kk & 1;
        bool more = (kk + 1) < ksteps;

        // ---- issue next weight-slice + next-A global loads before compute ----
        v8h w0, w1, w2, w3;
        V16 A0n, A1n;
        if (more) {
            const v8h* sp = (const v8h*)(wsrc + (kk + 1) * 32);
            w0 = sp[0]; w1 = sp[1]; w2 = sp[2]; w3 = sp[3];
            const half_t* a0p = arow0 + (kk + 1) * 32 + 8 * hf;
            const half_t* a1p = arow1 + (kk + 1) * 32 + 8 * hf;
            A0n.h[0] = *(const v8h*)a0p; A0n.h[1] = *(const v8h*)(a0p + 16);
            A1n.h[0] = *(const v8h*)a1p; A1n.h[1] = *(const v8h*)(a1p + 16);
        }

        // ---- load all 4 B fragments, then run 8 WMMAs back-to-back ----
        const half_t* lbase = &lw[buf][0];
        V16 Bv[4];
#pragma unroll
        for (int t = 0; t < 4; ++t) {
            const half_t* bp = lbase + (nq * 64 + t * 16 + m16) * 32 + 8 * hf;
            Bv[t].h[0] = *(const v8h*)bp;
            Bv[t].h[1] = *(const v8h*)(bp + 16);
        }
#pragma unroll
        for (int t = 0; t < 4; ++t) {
            acc[0][t] = __builtin_amdgcn_wmma_f32_16x16x32_f16(
                false, A0.v, false, Bv[t].v, (short)0, acc[0][t], false, false);
            acc[1][t] = __builtin_amdgcn_wmma_f32_16x16x32_f16(
                false, A1.v, false, Bv[t].v, (short)0, acc[1][t], false, false);
        }

        // ---- commit next slice to the other LDS buffer ----
        if (more) {
            v8h* dp = (v8h*)&lw[buf ^ 1][tid * 32];
            dp[0] = w0; dp[1] = w1; dp[2] = w2; dp[3] = w3;
            A0 = A0n; A1 = A1n;
        }
        __syncthreads();
    }

    // ---- epilogue: bias + ReLU into LDS stage, then coalesced writeout ----
    half_t* stage = &lw[0][0];                  // 64 rows x 256 cols f16 = 32 KB
#pragma unroll
    for (int st = 0; st < 2; ++st) {
#pragma unroll
        for (int t = 0; t < 4; ++t) {
            int ncol = nq * 64 + t * 16 + m16;
            float bv = bias[ncol];
#pragma unroll
            for (int v = 0; v < 8; ++v) {
                int rloc = strip * 32 + st * 16 + 8 * hf + v;
                float val = acc[st][t][v] + bv;
                val = fmaxf(val, 0.f);
                stage[rloc * HID + ncol] = (half_t)val;
            }
        }
    }
    __syncthreads();
    {
        int row = tid >> 2, seg = tid & 3;      // each thread moves 64 bytes
        const v8h* sp = (const v8h*)&stage[tid * 64];
        v8h* dp = (v8h*)&Y[(size_t)(blockIdx.x * 64 + row) * HID + seg * 64];
        dp[0] = sp[0]; dp[1] = sp[1]; dp[2] = sp[2]; dp[3] = sp[3];
    }
}

// ---------------- final 256->3 layer + weighted blend into d_out ------------
__global__ __launch_bounds__(256)
void final_kernel(const half_t* __restrict__ act, const float* __restrict__ W4,
                  const float* __restrict__ b4, const float* __restrict__ wgt,
                  float* __restrict__ out, int qc, int s) {
    int r = blockIdx.x * blockDim.x + threadIdx.x;
    if (r >= CHUNK) return;
    int n = qc * CHUNK + r;
    float a0 = b4[0], a1 = b4[1], a2 = b4[2];
    const half_t* ar = act + (size_t)r * HID;
#pragma unroll 4
    for (int k = 0; k < HID; ++k) {
        float a = (float)ar[k];
        a0 += a * W4[k * 3 + 0];
        a1 += a * W4[k * 3 + 1];
        a2 += a * W4[k * 3 + 2];
    }
    float w = wgt[(size_t)s * NQ + n];
    float* o = out + (size_t)n * 3;
    if (s == 0) { o[0] = w * a0; o[1] = w * a1; o[2] = w * a2; }
    else        { o[0] += w * a0; o[1] += w * a1; o[2] += w * a2; }
}

// ---------------- host launcher ---------------------------------------------
extern "C" void kernel_launch(void* const* d_in, const int* in_sizes, int n_in,
                              void* d_out, int out_size, void* d_ws, size_t ws_size,
                              hipStream_t stream) {
    (void)in_sizes; (void)n_in; (void)out_size; (void)ws_size;
    const float* feat  = (const float*)d_in[0];
    const float* lrf   = (const float*)d_in[1];
    const float* coord = (const float*)d_in[2];
    const float* cell  = (const float*)d_in[3];
    const float* W0 = (const float*)d_in[4];  const float* b0 = (const float*)d_in[5];
    const float* W1 = (const float*)d_in[6];  const float* b1 = (const float*)d_in[7];
    const float* W2 = (const float*)d_in[8];  const float* b2 = (const float*)d_in[9];
    const float* W3 = (const float*)d_in[10]; const float* b3 = (const float*)d_in[11];
    const float* W4 = (const float*)d_in[12]; const float* b4 = (const float*)d_in[13];
    float* out = (float*)d_out;

    char* ws = (char*)d_ws;
    half_t* WT0 = (half_t*)(ws + OFF_WT0);
    half_t* WT1 = (half_t*)(ws + OFF_WT1);
    half_t* WT2 = (half_t*)(ws + OFF_WT2);
    half_t* WT3 = (half_t*)(ws + OFF_WT3);
    float*  wgt = (float*)(ws + OFF_WGT);
    half_t* Xin = (half_t*)(ws + OFF_XIN);
    half_t* actA = (half_t*)(ws + OFF_ACTA);
    half_t* actB = (half_t*)(ws + OFF_ACTB);

    wcvt_kernel<<<256, 256, 0, stream>>>(W0, WT0, K0ACT, K0PAD);
    wcvt_kernel<<<256, 256, 0, stream>>>(W1, WT1, 256, 256);
    wcvt_kernel<<<256, 256, 0, stream>>>(W2, WT2, 256, 256);
    wcvt_kernel<<<256, 256, 0, stream>>>(W3, WT3, 256, 256);
    wgt_kernel<<<NQ / 256, 256, 0, stream>>>(coord, wgt);

    for (int s = 0; s < 4; ++s) {
        for (int qc = 0; qc < NCHK; ++qc) {
            build_kernel<<<CHUNK / 8, 256, 0, stream>>>(feat, lrf, coord, cell,
                                                        Xin, qc, s);
            gemm_kernel<<<CHUNK / 64, 256, 0, stream>>>(Xin,  WT0, b0, actA, K0PAD, K0PAD / 32);
            gemm_kernel<<<CHUNK / 64, 256, 0, stream>>>(actA, WT1, b1, actB, 256, 8);
            gemm_kernel<<<CHUNK / 64, 256, 0, stream>>>(actB, WT2, b2, actA, 256, 8);
            gemm_kernel<<<CHUNK / 64, 256, 0, stream>>>(actA, WT3, b3, actB, 256, 8);
            final_kernel<<<CHUNK / 256, 256, 0, stream>>>(actB, W4, b4, wgt, out, qc, s);
        }
    }
}
